// BMBC_68891275428427
// MI455X (gfx1250) — compile-verified, matrix operand order checked by reading, not run on hardware
//
#include <hip/hip_runtime.h>

// ---------------------------------------------------------------------------
// BMBC frame interpolation for MI455X (gfx1250, wave32, WMMA + TDM).
// Heavy convs run as implicit GEMM on v_wmma_f32_16x16x32_f16 with HWC-f16
// repacked activations (channel-padded to 32, zero halo => free conv padding)
// and [tap][oc][icp] repacked f16 weights. The dominant 408->32 conv streams
// its per-tap weight slab into LDS with the Tensor Data Mover (double-
// buffered, TENSORcnt-synchronized), so the hot loop is ds_load_b128 (B) +
// global_load_b128 (A) + v_wmma.
// ---------------------------------------------------------------------------

typedef __attribute__((ext_vector_type(16))) _Float16 v16h;
typedef __attribute__((ext_vector_type(8)))  float    v8f;

#define DI __device__ __forceinline__

constexpr int H   = 360, W = 640, HW = H * W;      // full res
constexpr int RH  = 384, RW = 640, RHW = RH * RW;  // bmnet res (ceil/32)
constexpr int PH  = H + 2, PW = W + 2;             // 3x3 conv padded extent
constexpr int CIN = 416;                           // 408 -> 13*32
constexpr int XC  = 68;                            // 67 -> 68 (I + ctx feat)
constexpr int CPH = H + 6, CPW = W + 6;            // 7x7 conv padded extent
constexpr int TAPW = 32 * CIN;                     // halves per df1 tap slab

DI int   imin(int a, int b) { return a < b ? a : b; }
DI int   imax(int a, int b) { return a > b ? a : b; }
DI float relu(float v) { return fmaxf(v, 0.f); }

DI v16h load16_a32(const _Float16* p) { return *reinterpret_cast<const v16h*>(p); }

DI v16h load16_a8(const _Float16* p) {  // 8B-aligned 32B load (ctx conv A frags)
  union { uint2 q[4]; v16h h; } u;
  const uint2* s = reinterpret_cast<const uint2*>(p);
  u.q[0] = s[0]; u.q[1] = s[1]; u.q[2] = s[2]; u.q[3] = s[3];
  return u.h;
}

// ----------------------- Tensor Data Mover (gfx1250) -----------------------
#if __has_builtin(__builtin_amdgcn_tensor_load_to_lds)
#define HAVE_TDM 1
typedef __attribute__((ext_vector_type(4))) unsigned int v4u;
typedef __attribute__((ext_vector_type(8))) int v8i;
typedef __attribute__((ext_vector_type(4))) int v4i;

// 1D slab copy global->LDS, nelem_h f16 elements. D# per CDNA5 ISA 8.3/8.4:
// group0: count=1 | lds_addr | global_addr[56:0] | type=2
// group1: data_size=2B, tensor_dim0=tile_dim0=nelem_h, tensor_dim1=tile_dim1=1
DI void tdm_load_slab(const void* gsrc, unsigned lds_byte_off, int nelem_h) {
  unsigned long long ga = (unsigned long long)(uintptr_t)gsrc;
  v4u g0;
  g0.x = 1u;                                          // count=1, is_restore=0
  g0.y = lds_byte_off;                                // D#.lds_addr
  g0.z = (unsigned)(ga & 0xffffffffull);              // global_addr[31:0]
  g0.w = (unsigned)((ga >> 32) & 0x01ffffffull) | (2u << 30);  // [56:32]|type=2
  v8i g1;
  g1[0] = 0x00010000;                                 // data_size=1 (2 bytes)
  g1[1] = (nelem_h & 0xffff) << 16;                   // tensor_dim0[15:0]
  g1[2] = (int)(((unsigned)nelem_h >> 16) | (1u << 16)); // dim0[31:16]|dim1=1
  g1[3] = (nelem_h & 0xffff) << 16;                   // tile_dim0 = nelem_h
  g1[4] = 1;                                          // tile_dim1=1, tile_dim2=0
  g1[5] = nelem_h;                                    // tensor_dim0_stride
  g1[6] = 0; g1[7] = 0;
  v4i z4 = (v4i){0, 0, 0, 0};
#if __has_include(<hip/amd_detail/amd_gfx1250_TDM.h>)
  v8i z8 = (v8i){0, 0, 0, 0, 0, 0, 0, 0};
  __builtin_amdgcn_tensor_load_to_lds(g0, g1, z4, z4, z8, 0);   // clang-23 lane
#else
  __builtin_amdgcn_tensor_load_to_lds(g0, g1, z4, z4, 0);       // ROCm 7.2 lane
#endif
}
#else
#define HAVE_TDM 0
#endif

// ------------------------------ utility ------------------------------------
__global__ void k_zero16(uint4* p, long n16) {
  long i = (long)blockIdx.x * blockDim.x + threadIdx.x;
  if (i < n16) p[i] = make_uint4(0u, 0u, 0u, 0u);
}

// tgt[0]=I0, tgt[1] written later, tgt[2]=target[2], optional count=2
__global__ void k_init_out(float* out, const float* I0, const float* target, int out_size) {
  int i = blockIdx.x * blockDim.x + threadIdx.x;
  if (i >= out_size) return;
  if (i < 3 * HW)       out[i] = I0[i];
  else if (i < 9 * HW)  out[i] = target[i];
  else                  out[i] = 2.0f;   // count = 2 after one intermediate frame
}

// ------------------- resize concat(I0,I1) -> pair_r (384x640) --------------
__global__ void k_resize_pair(const float* __restrict__ I0, const float* __restrict__ I1,
                              float* __restrict__ pair) {
  int p = blockIdx.x * blockDim.x + threadIdx.x;
  if (p >= RHW) return;
  int y = p / RW, x = p % RW;
  float sy = y * ((float)(H - 1) / (float)(RH - 1));
  float sx = x * ((float)(W - 1) / (float)(RW - 1));
  int y0 = (int)sy, x0 = (int)sx;
  int y1 = imin(y0 + 1, H - 1), x1 = imin(x0 + 1, W - 1);
  float wy = sy - y0, wx = sx - x0;
#pragma unroll
  for (int c = 0; c < 6; ++c) {
    const float* im = (c < 3) ? (I0 + c * HW) : (I1 + (c - 3) * HW);
    float v00 = im[y0 * W + x0], v01 = im[y0 * W + x1];
    float v10 = im[y1 * W + x0], v11 = im[y1 * W + x1];
    float r0 = v00 * (1.f - wx) + v01 * wx;
    float r1 = v10 * (1.f - wx) + v11 * wx;
    pair[c * RHW + p] = r0 * (1.f - wy) + r1 * wy;
  }
}

// ------------------- bmnet conv1: (6ch + t-ch) -> 32, 3x3, relu ------------
__global__ void k_bm_conv1(const float* __restrict__ pair, const float* __restrict__ w1,
                           const float* __restrict__ b1, float* __restrict__ hout, float t) {
  int p = blockIdx.x * blockDim.x + threadIdx.x;
  if (p >= RHW) return;
  int y = p / RW, x = p % RW;
  float acc[32];
#pragma unroll
  for (int o = 0; o < 32; ++o) acc[o] = b1[o];
  for (int ky = 0; ky < 3; ++ky) {
    int yy = y + ky - 1;
    for (int kx = 0; kx < 3; ++kx) {
      int xx = x + kx - 1;
      bool inb = (yy >= 0) & (yy < RH) & (xx >= 0) & (xx < RW);
      for (int ic = 0; ic < 7; ++ic) {
        float v = inb ? (ic < 6 ? pair[ic * RHW + yy * RW + xx] : t) : 0.f;
        const float* wp = w1 + (ic * 3 + ky) * 3 + kx;   // uniform -> s_load
#pragma unroll
        for (int o = 0; o < 32; ++o) acc[o] += v * wp[o * 63];
      }
    }
  }
#pragma unroll
  for (int o = 0; o < 32; ++o) hout[o * RHW + p] = relu(acc[o]);
}

// ------------------- bmnet conv2: 32 -> 2, 3x3 -----------------------------
__global__ void k_bm_conv2(const float* __restrict__ hin, const float* __restrict__ w2,
                           const float* __restrict__ b2, float* __restrict__ bmo) {
  int p = blockIdx.x * blockDim.x + threadIdx.x;
  if (p >= RHW) return;
  int y = p / RW, x = p % RW;
  float a0 = b2[0], a1 = b2[1];
  for (int ky = 0; ky < 3; ++ky) {
    int yy = y + ky - 1;
    for (int kx = 0; kx < 3; ++kx) {
      int xx = x + kx - 1;
      bool inb = (yy >= 0) & (yy < RH) & (xx >= 0) & (xx < RW);
      if (!inb) continue;
#pragma unroll 8
      for (int ic = 0; ic < 32; ++ic) {
        float v = hin[ic * RHW + yy * RW + xx];
        a0 += v * w2[(ic * 3 + ky) * 3 + kx];
        a1 += v * w2[((32 + ic) * 3 + ky) * 3 + kx];
      }
    }
  }
  bmo[p] = a0; bmo[RHW + p] = a1;
}

// ------------- resize 384x640 -> 360x640 with per-channel scale ------------
__global__ void k_resize_flow(const float* __restrict__ bmo, float* __restrict__ flow,
                              float mulx, float muly) {
  int p = blockIdx.x * blockDim.x + threadIdx.x;
  if (p >= HW) return;
  int y = p / W, x = p % W;
  float sy = y * ((float)(RH - 1) / (float)(H - 1));
  float sx = x * ((float)(RW - 1) / (float)(W - 1));
  int y0 = (int)sy, x0 = (int)sx;
  int y1 = imin(y0 + 1, RH - 1), x1 = imin(x0 + 1, RW - 1);
  float wy = sy - y0, wx = sx - x0;
#pragma unroll
  for (int c = 0; c < 2; ++c) {
    const float* im = bmo + c * RHW;
    float v00 = im[y0 * RW + x0], v01 = im[y0 * RW + x1];
    float v10 = im[y1 * RW + x0], v11 = im[y1 * RW + x1];
    float v = (v00 * (1.f - wx) + v01 * wx) * (1.f - wy) + (v10 * (1.f - wx) + v11 * wx) * wy;
    flow[c * HW + p] = v * (c == 0 ? mulx : muly);
  }
}

// ------------- pack image into padded HWC f16 [CPH][CPW][4] ----------------
__global__ void k_pack_img(const float* __restrict__ I, _Float16* __restrict__ dst) {
  int p = blockIdx.x * blockDim.x + threadIdx.x;
  if (p >= CPH * CPW) return;
  int py = p / CPW, px = p % CPW;
  int y = py - 3, x = px - 3;
  bool inb = (y >= 0) & (y < H) & (x >= 0) & (x < W);
  _Float16* d = dst + (size_t)p * 4;
  d[0] = inb ? (_Float16)I[0 * HW + y * W + x] : (_Float16)0.f;
  d[1] = inb ? (_Float16)I[1 * HW + y * W + x] : (_Float16)0.f;
  d[2] = inb ? (_Float16)I[2 * HW + y * W + x] : (_Float16)0.f;
  d[3] = (_Float16)0.f;
}

// X[y][x][0..2] = I, X[y][x][67] = 0   (ctx conv fills 3..66)
__global__ void k_x_base(const float* __restrict__ I, _Float16* __restrict__ X) {
  int p = blockIdx.x * blockDim.x + threadIdx.x;
  if (p >= HW) return;
  _Float16* d = X + (size_t)p * XC;
  d[0] = (_Float16)I[p];
  d[1] = (_Float16)I[HW + p];
  d[2] = (_Float16)I[2 * HW + p];
  d[67] = (_Float16)0.f;
}

// ctx_W (64,3,7,7) -> wctx[o][dy][k], k = dx*4 + c (28 valid of 32, pads = 0)
__global__ void k_pack_wctx(const float* __restrict__ wsrc, _Float16* __restrict__ wd) {
  int i = blockIdx.x * blockDim.x + threadIdx.x;
  if (i >= 64 * 7 * 32) return;
  int o = i / (7 * 32), dy = (i / 32) % 7, k = i % 32;
  int dx = k >> 2, c = k & 3;
  float v = (dx < 7 && c < 3) ? wsrc[((o * 3 + c) * 7 + dy) * 7 + dx] : 0.f;
  wd[i] = (_Float16)v;
}

// ---------------- ctx 7x7 conv via WMMA: 16 px x 64 oc per wave ------------
__global__ __launch_bounds__(256) void k_ctx_wmma(const _Float16* __restrict__ src,
                                                  const _Float16* __restrict__ wctx,
                                                  _Float16* __restrict__ X) {
  int lane = threadIdx.x & 31;
  int gw   = blockIdx.x * 8 + (threadIdx.x >> 5);
  int pb   = gw * 16;
  int m = lane & 15, kh = lane >> 4;
  int p = pb + m, y = p / W, x = p % W;
  v8f acc[4];
#pragma unroll
  for (int nt = 0; nt < 4; ++nt) acc[nt] = (v8f){};
#pragma unroll
  for (int dy = 0; dy < 7; ++dy) {
    v16h a = load16_a8(src + ((size_t)(y + dy) * CPW + x) * 4 + kh * 16);
    const _Float16* bp = wctx + ((size_t)m * 7 + dy) * 32 + kh * 16;
    v16h b0 = load16_a32(bp);
    v16h b1 = load16_a32(bp + 16 * 7 * 32);
    v16h b2 = load16_a32(bp + 32 * 7 * 32);
    v16h b3 = load16_a32(bp + 48 * 7 * 32);
    acc[0] = __builtin_amdgcn_wmma_f32_16x16x32_f16(false, a, false, b0, (short)0, acc[0], false, false);
    acc[1] = __builtin_amdgcn_wmma_f32_16x16x32_f16(false, a, false, b1, (short)0, acc[1], false, false);
    acc[2] = __builtin_amdgcn_wmma_f32_16x16x32_f16(false, a, false, b2, (short)0, acc[2], false, false);
    acc[3] = __builtin_amdgcn_wmma_f32_16x16x32_f16(false, a, false, b3, (short)0, acc[3], false, false);
  }
#pragma unroll
  for (int nt = 0; nt < 4; ++nt) {
    int ch = 3 + nt * 16 + m;
#pragma unroll
    for (int r = 0; r < 8; ++r) {
      int p2 = pb + r + 8 * kh;
      X[(size_t)p2 * XC + ch] = (_Float16)relu(acc[nt][r]);
    }
  }
}

// ---------------- fused 6-way warp + assemble inp [PH][PW][CIN] ------------
__global__ void k_warp_assemble(const _Float16* __restrict__ X0, const _Float16* __restrict__ X1,
                                const float* __restrict__ F01, const float* __restrict__ F10,
                                const float* __restrict__ BM,
                                const float* __restrict__ I0, const float* __restrict__ I1,
                                _Float16* __restrict__ inp) {
  int p = blockIdx.x * blockDim.x + threadIdx.x;
  if (p >= HW) return;
  int y = p / W, x = p % W;
  _Float16* dst = inp + ((size_t)(y + 1) * PW + (x + 1)) * CIN;
  dst[0] = (_Float16)I0[p]; dst[1] = (_Float16)I0[HW + p]; dst[2] = (_Float16)I0[2 * HW + p];
  dst[405] = (_Float16)I1[p]; dst[406] = (_Float16)I1[HW + p]; dst[407] = (_Float16)I1[2 * HW + p];

  float fx6[6], fy6[6];
  float ax = F01[p], ay = F01[HW + p];
  float bx = F10[p], by = F10[HW + p];
  float mx = BM[p],  my = BM[HW + p];
  fx6[0] = -0.5f * ax; fy6[0] = -0.5f * ay;   // C1 = warp(X0, -t*F01)
  fx6[1] =  0.5f * ax; fy6[1] =  0.5f * ay;   // C2 = warp(X1, ti*F01)
  fx6[2] =  0.5f * bx; fy6[2] =  0.5f * by;   // C3 = warp(X0, t*F10)
  fx6[3] = -0.5f * bx; fy6[3] = -0.5f * by;   // C4 = warp(X1, (t-1)*F10)
  fx6[4] = -1.0f * mx; fy6[4] = -1.0f * my;   // C5 = warp(X0, -2t*BM)
  fx6[5] =  1.0f * mx; fy6[5] =  1.0f * my;   // C6 = warp(X1, 2ti*BM)

  for (int cd = 0; cd < 6; ++cd) {
    const _Float16* img = (cd & 1) ? X1 : X0;
    float xq = (float)x + fx6[cd], yq = (float)y + fy6[cd];
    int x0 = (int)floorf(xq), y0 = (int)floorf(yq);
    int x1 = x0 + 1, y1 = y0 + 1;
    float wx = xq - x0, wy = yq - y0;
    float w00 = (1.f - wx) * (1.f - wy), w10 = wx * (1.f - wy);
    float w01 = (1.f - wx) * wy,         w11 = wx * wy;
    bool vx0 = (x0 >= 0) & (x0 < W), vx1 = (x1 >= 0) & (x1 < W);
    bool vy0 = (y0 >= 0) & (y0 < H), vy1 = (y1 >= 0) & (y1 < H);
    if (!(vx0 & vy0)) w00 = 0.f;
    if (!(vx1 & vy0)) w10 = 0.f;
    if (!(vx0 & vy1)) w01 = 0.f;
    if (!(vx1 & vy1)) w11 = 0.f;
    int xc0 = imin(imax(x0, 0), W - 1), xc1 = imin(imax(x1, 0), W - 1);
    int yc0 = imin(imax(y0, 0), H - 1), yc1 = imin(imax(y1, 0), H - 1);
    const _Float16* p00 = img + ((size_t)yc0 * W + xc0) * XC;
    const _Float16* p10 = img + ((size_t)yc0 * W + xc1) * XC;
    const _Float16* p01 = img + ((size_t)yc1 * W + xc0) * XC;
    const _Float16* p11 = img + ((size_t)yc1 * W + xc1) * XC;
    _Float16* d = dst + 3 + cd * 67;
    for (int c = 0; c < 67; ++c) {
      float v = w00 * (float)p00[c] + w10 * (float)p10[c] +
                w01 * (float)p01[c] + w11 * (float)p11[c];
      d[c] = (_Float16)v;
    }
  }
}

// df_W1 (32,408,3,3) -> wdf1[tap][o][cip] f16 (408->416, pads 0); one
// contiguous 26.6KB slab per tap => single TDM descriptor per prefetch.
__global__ void k_pack_wdf1(const float* __restrict__ wsrc, _Float16* __restrict__ wd) {
  int i = blockIdx.x * blockDim.x + threadIdx.x;
  if (i >= 9 * 32 * CIN) return;
  int tap = i / (32 * CIN), o = (i / CIN) % 32, c = i % CIN;
  float v = (c < 408) ? wsrc[((o * 408 + c) * 3 + tap / 3) * 3 + tap % 3] : 0.f;
  wd[i] = (_Float16)v;
}

// --------------- df conv1 WMMA: 16 px x 32 oc, K = 9 taps * 416 ------------
// Per-tap weight slab staged in LDS by the Tensor Data Mover, double-buffered:
// prefetch tap+1 while computing tap, rotate on s_wait_tensorcnt + barrier.
__global__ __launch_bounds__(256) void k_df1_wmma(const _Float16* __restrict__ inp,
                                                  const _Float16* __restrict__ wdf1,
                                                  const float* __restrict__ bias,
                                                  _Float16* __restrict__ hh) {
  __shared__ _Float16 sw[2][TAPW];   // 2 x 26.6 KB (only LDS object: offset 0)
  const int tid = threadIdx.x;
  int lane = tid & 31;
  int wv   = tid >> 5;
  int gw   = blockIdx.x * 8 + wv;
  int pb   = gw * 16;
  int m = lane & 15, kh = lane >> 4;
  int p = pb + m, y = p / W, x = p % W;

#if HAVE_TDM
  if (wv == 0) {                       // one wave drives the DMA engine
    tdm_load_slab(wdf1, 0u, TAPW);
    __builtin_amdgcn_s_wait_tensorcnt((short)0);
  }
  __syncthreads();
#else
  for (int i = tid; i < TAPW / 8; i += 256)
    ((uint4*)sw[0])[i] = ((const uint4*)wdf1)[i];
  __syncthreads();
#endif

  v8f acc0 = (v8f){}, acc1 = (v8f){};
  int cur = 0;
  for (int tap = 0; tap < 9; ++tap) {
#if HAVE_TDM
    if (tap < 8 && wv == 0)            // async prefetch of next tap slab
      tdm_load_slab(wdf1 + (size_t)(tap + 1) * TAPW,
                    (unsigned)((1 - cur) * TAPW * 2), TAPW);
#endif
    int dy = tap / 3, dx = tap % 3;
    const _Float16* abase = inp + ((size_t)(y + dy) * PW + (x + dx)) * CIN + kh * 16;
    const _Float16* bbase = sw[cur] + (size_t)m * CIN + kh * 16;
#pragma unroll
    for (int ck = 0; ck < 13; ++ck) {
      v16h a  = load16_a32(abase + ck * 32);
      v16h b0 = *(const v16h*)(bbase + ck * 32);
      v16h b1 = *(const v16h*)(bbase + 16 * CIN + ck * 32);
      acc0 = __builtin_amdgcn_wmma_f32_16x16x32_f16(false, a, false, b0, (short)0, acc0, false, false);
      acc1 = __builtin_amdgcn_wmma_f32_16x16x32_f16(false, a, false, b1, (short)0, acc1, false, false);
    }
    if (tap < 8) {
#if HAVE_TDM
      if (wv == 0) __builtin_amdgcn_s_wait_tensorcnt((short)0);
      __syncthreads();
#else
      __syncthreads();
      for (int i = tid; i < TAPW / 8; i += 256)
        ((uint4*)sw[1 - cur])[i] = ((const uint4*)(wdf1 + (size_t)(tap + 1) * TAPW))[i];
      __syncthreads();
#endif
      cur ^= 1;
    }
  }

  float bb0 = bias[m], bb1 = bias[16 + m];
#pragma unroll
  for (int r = 0; r < 8; ++r) {
    int p2 = pb + r + 8 * kh;
    int y2 = p2 / W, x2 = p2 % W;
    _Float16* d = hh + ((size_t)(y2 + 1) * PW + (x2 + 1)) * 32;
    d[m]      = (_Float16)relu(acc0[r] + bb0);
    d[16 + m] = (_Float16)relu(acc1[r] + bb1);
  }
}

// df_W2 (54,32,3,3) -> wdf2[o<64][tap][32] f16 (54->64, pads 0)
__global__ void k_pack_wdf2(const float* __restrict__ wsrc, _Float16* __restrict__ wd) {
  int i = blockIdx.x * blockDim.x + threadIdx.x;
  if (i >= 64 * 9 * 32) return;
  int o = i / (9 * 32), tap = (i / 32) % 9, c = i % 32;
  float v = (o < 54) ? wsrc[((o * 32 + c) * 3 + tap / 3) * 3 + tap % 3] : 0.f;
  wd[i] = (_Float16)v;
}

// ---- fused: df conv2 WMMA -> LDS -> per-pixel softmax -> dynfilter --------
constexpr int SMS = 66;   // padded row stride (words) => conflict-free reads

__global__ __launch_bounds__(256) void k_df2_fused(const _Float16* __restrict__ hh,
                                                   const _Float16* __restrict__ wdf2,
                                                   const float* __restrict__ bias,
                                                   const _Float16* __restrict__ inp,
                                                   float* __restrict__ out) {
  __shared__ float sm[128 * SMS];
  int lane = threadIdx.x & 31;
  int wv   = threadIdx.x >> 5;
  int pb   = blockIdx.x * 128 + wv * 16;
  int m = lane & 15, kh = lane >> 4;
  int p = pb + m, y = p / W, x = p % W;
  v8f acc[4];
#pragma unroll
  for (int nt = 0; nt < 4; ++nt) acc[nt] = (v8f){};
#pragma unroll
  for (int tap = 0; tap < 9; ++tap) {
    int dy = tap / 3, dx = tap % 3;
    v16h a = load16_a32(hh + ((size_t)(y + dy) * PW + (x + dx)) * 32 + kh * 16);
    const _Float16* bp = wdf2 + ((size_t)m * 9 + tap) * 32 + kh * 16;
    v16h b0 = load16_a32(bp);
    v16h b1 = load16_a32(bp + 16 * 9 * 32);
    v16h b2 = load16_a32(bp + 32 * 9 * 32);
    v16h b3 = load16_a32(bp + 48 * 9 * 32);
    acc[0] = __builtin_amdgcn_wmma_f32_16x16x32_f16(false, a, false, b0, (short)0, acc[0], false, false);
    acc[1] = __builtin_amdgcn_wmma_f32_16x16x32_f16(false, a, false, b1, (short)0, acc[1], false, false);
    acc[2] = __builtin_amdgcn_wmma_f32_16x16x32_f16(false, a, false, b2, (short)0, acc[2], false, false);
    acc[3] = __builtin_amdgcn_wmma_f32_16x16x32_f16(false, a, false, b3, (short)0, acc[3], false, false);
  }
#pragma unroll
  for (int nt = 0; nt < 4; ++nt) {
    int ch = nt * 16 + m;
    float b = (ch < 54) ? bias[ch] : 0.f;
#pragma unroll
    for (int r = 0; r < 8; ++r) {
      int pl = wv * 16 + r + 8 * kh;
      sm[pl * SMS + ch] = acc[nt][r] + b;
    }
  }
  __syncthreads();

  int tid = threadIdx.x;
  if (tid < 128) {
    int pg = blockIdx.x * 128 + tid;
    int yy = pg / W, xx = pg % W;
    float lg[54];
    float mx = -3.4e38f;
#pragma unroll
    for (int k = 0; k < 54; ++k) { lg[k] = sm[tid * SMS + k]; mx = fmaxf(mx, lg[k]); }
    float s = 0.f;
#pragma unroll
    for (int k = 0; k < 54; ++k) { lg[k] = __expf(lg[k] - mx); s += lg[k]; }
    float inv = 1.f / s;
    float R = 0.f, G = 0.f, Bc = 0.f;
    for (int c6 = 0; c6 < 6; ++c6) {
      int cb = 3 + 67 * c6;
#pragma unroll
      for (int tap = 0; tap < 9; ++tap) {
        float wgt = lg[c6 * 9 + tap];
        const _Float16* q = inp + ((size_t)(yy + tap / 3) * PW + (xx + tap % 3)) * CIN + cb;
        R += wgt * (float)q[0]; G += wgt * (float)q[1]; Bc += wgt * (float)q[2];
      }
    }
    out[3 * HW + pg]          = R * inv;   // frame 1, channel R
    out[3 * HW + HW + pg]     = G * inv;
    out[3 * HW + 2 * HW + pg] = Bc * inv;
  }
}

// ---------------------------------------------------------------------------
extern "C" void kernel_launch(void* const* d_in, const int* in_sizes, int n_in,
                              void* d_out, int out_size, void* d_ws, size_t ws_size,
                              hipStream_t stream) {
  (void)in_sizes; (void)n_in; (void)ws_size;
  const float* I0    = (const float*)d_in[0];
  const float* I1    = (const float*)d_in[1];
  const float* tgt   = (const float*)d_in[2];
  const float* ctxW  = (const float*)d_in[3];
  const float* bmW1  = (const float*)d_in[4];
  const float* bmB1  = (const float*)d_in[5];
  const float* bmW2  = (const float*)d_in[6];
  const float* bmB2  = (const float*)d_in[7];
  const float* dfW1  = (const float*)d_in[8];
  const float* dfB1  = (const float*)d_in[9];
  const float* dfW2  = (const float*)d_in[10];
  const float* dfB2  = (const float*)d_in[11];
  float* out = (float*)d_out;

  char* ws = (char*)d_ws;
  size_t off = 0;
  auto alloc = [&](size_t bytes) { size_t r = off; off = (off + bytes + 255) & ~(size_t)255; return r; };

  float*    pair = (float*)(ws + alloc((size_t)6 * RHW * 4));
  float*    bmh  = (float*)(ws + alloc((size_t)32 * RHW * 4));
  float*    bmo  = (float*)(ws + alloc((size_t)2 * RHW * 4));
  float*    F01  = (float*)(ws + alloc((size_t)2 * HW * 4));
  float*    F10  = (float*)(ws + alloc((size_t)2 * HW * 4));
  float*    BM   = (float*)(ws + alloc((size_t)2 * HW * 4));
  _Float16* i0p  = (_Float16*)(ws + alloc(((size_t)CPH * CPW * 4 + 64) * 2));
  _Float16* i1p  = (_Float16*)(ws + alloc(((size_t)CPH * CPW * 4 + 64) * 2));
  _Float16* X0   = (_Float16*)(ws + alloc((size_t)HW * XC * 2));
  _Float16* X1   = (_Float16*)(ws + alloc((size_t)HW * XC * 2));
  _Float16* wctx = (_Float16*)(ws + alloc((size_t)64 * 7 * 32 * 2));
  _Float16* inp  = (_Float16*)(ws + alloc((size_t)PH * PW * CIN * 2));
  _Float16* wdf1 = (_Float16*)(ws + alloc((size_t)9 * 32 * CIN * 2));
  _Float16* hh   = (_Float16*)(ws + alloc((size_t)PH * PW * 32 * 2));
  _Float16* wdf2 = (_Float16*)(ws + alloc((size_t)64 * 9 * 32 * 2));

  const int T = 256;

  // zero the two halo-padded f16 activation buffers (conv zero-padding)
  long n16a = (long)PH * PW * CIN / 8;
  long n16b = (long)PH * PW * 32 / 8;
  k_zero16<<<(int)((n16a + T - 1) / T), T, 0, stream>>>((uint4*)inp, n16a);
  k_zero16<<<(int)((n16b + T - 1) / T), T, 0, stream>>>((uint4*)hh, n16b);

  k_init_out<<<(out_size + T - 1) / T, T, 0, stream>>>(out, I0, tgt, out_size);

  // bmnet x3 -> flows
  k_resize_pair<<<(RHW + T - 1) / T, T, 0, stream>>>(I0, I1, pair);
  const float scly = (float)H / (float)RH;              // 0.9375 ; sclx = 1.0
  struct { float t, mx, my; float* dst; } runs[3] = {
    {0.0f,  2.0f,  2.0f * scly, F01},
    {1.0f, -2.0f, -2.0f * scly, F10},
    {0.5f,  1.0f,  1.0f * scly, BM },
  };
  for (int r = 0; r < 3; ++r) {
    k_bm_conv1<<<(RHW + T - 1) / T, T, 0, stream>>>(pair, bmW1, bmB1, bmh, runs[r].t);
    k_bm_conv2<<<(RHW + T - 1) / T, T, 0, stream>>>(bmh, bmW2, bmB2, bmo);
    k_resize_flow<<<(HW + T - 1) / T, T, 0, stream>>>(bmo, runs[r].dst, runs[r].mx, runs[r].my);
  }

  // context features via WMMA
  k_pack_img<<<(CPH * CPW + T - 1) / T, T, 0, stream>>>(I0, i0p);
  k_pack_img<<<(CPH * CPW + T - 1) / T, T, 0, stream>>>(I1, i1p);
  k_x_base<<<(HW + T - 1) / T, T, 0, stream>>>(I0, X0);
  k_x_base<<<(HW + T - 1) / T, T, 0, stream>>>(I1, X1);
  k_pack_wctx<<<(64 * 7 * 32 + T - 1) / T, T, 0, stream>>>(ctxW, wctx);
  k_ctx_wmma<<<HW / 128, T, 0, stream>>>(i0p, wctx, X0);
  k_ctx_wmma<<<HW / 128, T, 0, stream>>>(i1p, wctx, X1);

  // warp + assemble 408-channel input
  k_warp_assemble<<<(HW + T - 1) / T, T, 0, stream>>>(X0, X1, F01, F10, BM, I0, I1, inp);

  // dynamic-filter convs via WMMA (TDM-staged weights) + fused softmax/filter
  k_pack_wdf1<<<(9 * 32 * CIN + T - 1) / T, T, 0, stream>>>(dfW1, wdf1);
  k_df1_wmma<<<HW / 128, T, 0, stream>>>(inp, wdf1, dfB1, hh);
  k_pack_wdf2<<<(64 * 9 * 32 + T - 1) / T, T, 0, stream>>>(dfW2, wdf2);
  k_df2_fused<<<HW / 128, T, 0, stream>>>(hh, wdf2, dfB2, inp, out);
}
